// MultiHead_29600914604040
// MI455X (gfx1250) — compile-verified, matrix operand order checked by reading, not run on hardware
//
#include <hip/hip_runtime.h>
#include <hip/hip_bf16.h>
#include <math.h>

// ---------------------------------------------------------------------------
// MultiHead attention for MI455X (gfx1250, wave32, WMMA).
//   B=4, T=2048, D=512, H=8, hd=64.
// Outputs (concatenated): context (B,T,D) f32, dist_mean (B,T,T) f32.
//
// Roofline: dist_mean write (67MB) + inputs (~50MB) -> ~6us @ 23.3TB/s.
// Compute ~52 GFLOP -> bf16 WMMA (v_wmma_f32_16x16x32_bf16); K/V (16MB)
// are L2-resident (192MB) so per-tile re-reads are free.
//
// All Q/K/V data is pre-packed into WMMA "fragment-major" layout: for each
// (b,h,tile,frag,lane) the 16 bf16 values that lane feeds to the WMMA are
// contiguous (32B), so the hot loop is pure global_load_b128 -> v_wmma.
// ---------------------------------------------------------------------------

typedef __attribute__((ext_vector_type(16))) __bf16 v16bf;
typedef __attribute__((ext_vector_type(8)))  float  v8f;

#define B_    4
#define T_    2048
#define D_    512
#define H_    8
#define HD_   64
#define QT_   16

static __device__ __forceinline__ v8f wmma_bf16(v16bf a, v16bf b, v8f c) {
    // D = A(16x32 bf16) * B(32x16 bf16) + C(16x16 f32)
    return __builtin_amdgcn_wmma_f32_16x16x32_bf16(
        /*neg_a=*/false, a, /*neg_b=*/false, b,
        /*c_mod=*/(short)0, c, /*reuse_a=*/false, /*reuse_b=*/false);
}

// --------------------------- pack kernels ----------------------------------
// Fragment element e (0..15) of a 16-bit WMMA A/B fragment maps to
// K-index: kblo + e + (e>=8 ? 8 : 0), kblo = (lane<16 ? 0 : 8)   (ISA 7.12.2)
// Row (A) / column (B) index = lane & 15.

// Q and K: per (b,h): 128 k-tiles x 2 frags (hd 0..31 / 32..63) x 32 lanes x 16.
__global__ __launch_bounds__(256)
void pack_frag_qk_kernel(const float* __restrict__ q, const float* __restrict__ k,
                         __bf16* __restrict__ qp, __bf16* __restrict__ kp) {
    const int o    = blockIdx.x * 256 + threadIdx.x;   // 0 .. B*H*128*2*512-1
    const int e    = o & 15;
    const int lane = (o >> 4) & 31;
    const int f    = (o >> 9) & 1;
    const int kt   = (o >> 10) & 127;
    const int bh   = o >> 17;
    const int row  = kt * 16 + (lane & 15);
    const int kblo = (lane < 16) ? 0 : 8;
    const int hd   = f * 32 + kblo + e + ((e >= 8) ? 8 : 0);
    const size_t src =
        ((size_t)((bh >> 3) * T_ + row)) * D_ + (bh & 7) * HD_ + hd;
    qp[o] = (__bf16)q[src];
    kp[o] = (__bf16)k[src];
}

// V: per (b,h): 64 k-steps (32 keys) x 4 n-tiles x 32 lanes x 16 elements.
// Element e maps along the KEY axis (GEMM2 K-dim); column = head-dim index.
__global__ __launch_bounds__(256)
void pack_frag_v_kernel(const float* __restrict__ v, __bf16* __restrict__ vp) {
    __shared__ float tile[32][64];                     // 32 keys x 64 head-dim
    const int blk = blockIdx.x;                        // (b*H + h)*64 + ks
    const int ks  = blk & 63;
    const int bh  = blk >> 6;
    const int b   = bh >> 3, h = bh & 7;
    const int tx  = threadIdx.x & 63;                  // head-dim n
    const int ty  = threadIdx.x >> 6;                  // 0..3
    const int k0  = ks * 32;
#pragma unroll
    for (int j = 0; j < 32; j += 4)
        tile[ty + j][tx] =
            v[((size_t)(b * T_ + k0 + ty + j)) * D_ + h * HD_ + tx];
    __syncthreads();
    const size_t obase = ((size_t)blk) * 2048 + threadIdx.x * 8;
#pragma unroll
    for (int i = 0; i < 8; ++i) {
        const int o    = threadIdx.x * 8 + i;          // 0..2047 within block
        const int e    = o & 15;
        const int lane = (o >> 4) & 31;
        const int nt   = (o >> 9) & 3;
        const int kblo = (lane < 16) ? 0 : 8;
        const int kk   = kblo + e + ((e >= 8) ? 8 : 0);
        vp[obase + i]  = (__bf16)tile[kk][nt * 16 + (lane & 15)];
    }
}

// --------------------------- attention kernel ------------------------------
// Block: 256 threads = 8 waves; wave w handles head w for a 16-row q tile.

__global__ __launch_bounds__(256)
void attn_kernel(const __bf16* __restrict__ qpk,
                 const __bf16* __restrict__ kpk,
                 const __bf16* __restrict__ vpk,
                 const int*    __restrict__ key_mask,
                 float*        __restrict__ ctx,    // (B,T,D)
                 float*        __restrict__ dist)   // (B,T,T)
{
    const int b    = blockIdx.y;
    const int qt   = blockIdx.x;
    const int q0   = qt * QT_;
    const int lane = threadIdx.x & 31;
    const int h    = threadIdx.x >> 5;          // wave id == head
    const int col  = lane & 15;
    const int rb   = (lane < 16) ? 0 : 8;       // C-tile row base for this lane
    const int kblo = (lane < 16) ? 0 : 8;

    __shared__ float Pbuf[H_][QT_][32];         // 16 KB: per-head prob tiles

    const float scale = 0.125f;                 // 1/sqrt(64)
    const float NEGS  = -1e20f * 0.125f;        // masked-then-scaled score

    // Fragment bases: each (b,h) owns 256 fragments of 32 lanes (v16bf each).
    const size_t bh = (size_t)(b * H_ + h);
    const v16bf* qf = (const v16bf*)qpk + bh * 256 * 32 + lane;
    const v16bf* kf = (const v16bf*)kpk + bh * 256 * 32 + lane;
    const v16bf* vf = (const v16bf*)vpk + bh * 256 * 32 + lane;
    const int*   mk = key_mask + b * T_;

    // ---- Q fragments (hd split 2x32), single vector loads ----
    const v16bf aq0 = qf[(qt * 2 + 0) * 32];
    const v16bf aq1 = qf[(qt * 2 + 1) * 32];

    // ---- Pass 1: lane-local online max/sum (no cross-lane work in loop) ----
    float m[8], l[8];
#pragma unroll
    for (int v = 0; v < 8; ++v) { m[v] = -3.0e38f; l[v] = 0.0f; }

    for (int kt = 0; kt < 128; ++kt) {
        v16bf bk0 = kf[(kt * 2 + 0) * 32];
        v16bf bk1 = kf[(kt * 2 + 1) * 32];
        v8f c = {};
        c = wmma_bf16(aq0, bk0, c);
        c = wmma_bf16(aq1, bk1, c);
        const bool mok = mk[kt * 16 + col] != 0;
#pragma unroll
        for (int v = 0; v < 8; ++v) {
            const float s  = mok ? c[v] * scale : NEGS;
            const float nm = fmaxf(m[v], s);
            // one exp per element: exp(min-max); the other factor is 1.
            const float em = __expf(fminf(m[v], s) - nm);
            const bool  sg = s > m[v];
            l[v] = l[v] * (sg ? em : 1.0f) + (sg ? 1.0f : em);
            m[v] = nm;
        }
    }

    // ---- one-time merge across the 16 lanes sharing each row ----
    float inv_l[8];
#pragma unroll
    for (int v = 0; v < 8; ++v) {
        float mt = m[v];
        mt = fmaxf(mt, __shfl_xor(mt, 1, 32));
        mt = fmaxf(mt, __shfl_xor(mt, 2, 32));
        mt = fmaxf(mt, __shfl_xor(mt, 4, 32));
        mt = fmaxf(mt, __shfl_xor(mt, 8, 32));
        float lt = l[v] * __expf(m[v] - mt);
        lt += __shfl_xor(lt, 1, 32);
        lt += __shfl_xor(lt, 2, 32);
        lt += __shfl_xor(lt, 4, 32);
        lt += __shfl_xor(lt, 8, 32);
        m[v]     = mt;
        inv_l[v] = 1.0f / lt;
    }

    // ---- Pass 2: P tiles -> dist_mean + O accumulation ----
    v8f o[4] = {};                              // O: 16x64 as four 16x16 tiles

    for (int ks = 0; ks < 64; ++ks) {
        const int k0 = ks * 32;
#pragma unroll
        for (int t = 0; t < 2; ++t) {
            const int kt = ks * 2 + t;
            v16bf bk0 = kf[(kt * 2 + 0) * 32];
            v16bf bk1 = kf[(kt * 2 + 1) * 32];
            v8f c = {};
            c = wmma_bf16(aq0, bk0, c);
            c = wmma_bf16(aq1, bk1, c);
            const bool mok = mk[kt * 16 + col] != 0;
#pragma unroll
            for (int v = 0; v < 8; ++v) {
                const float s = mok ? c[v] * scale : NEGS;
                const float p = __expf(s - m[v]) * inv_l[v];
                Pbuf[h][rb + v][t * 16 + col] = p;   // C-layout scatter to LDS
            }
        }
        __syncthreads();

        // dist_mean: 512 elements, 2 per thread, coalesced rows of 32.
        for (int e = threadIdx.x; e < 512; e += 256) {
            const int mm = e >> 5, cc = e & 31;
            float sum = 0.0f;
#pragma unroll
            for (int hh = 0; hh < H_; ++hh) sum += Pbuf[hh][mm][cc];
            dist[((size_t)(b * T_ + q0 + mm)) * T_ + k0 + cc] = sum * 0.125f;
        }

        // GEMM2: A = own P tile (LDS readback does the transpose), B = V frags.
        {
            const float* pp = &Pbuf[h][col][0];  // A-layout: row m = lane&15
            v16bf ap;
#pragma unroll
            for (int i = 0; i < 8; ++i) {
                ap[i]     = (__bf16)pp[kblo + i];
                ap[8 + i] = (__bf16)pp[kblo + 16 + i];
            }
#pragma unroll
            for (int nt = 0; nt < 4; ++nt) {
                v16bf bv = vf[(ks * 4 + nt) * 32];
                o[nt] = wmma_bf16(ap, bv, o[nt]);
            }
        }
        __syncthreads();                         // protect Pbuf reuse
    }

    // ---- store context ----
#pragma unroll
    for (int nt = 0; nt < 4; ++nt)
#pragma unroll
        for (int v = 0; v < 8; ++v)
            ctx[((size_t)(b * T_ + q0 + rb + v)) * D_ + h * HD_ + nt * 16 + col] =
                o[nt][v];
}

// --------------------------- launcher --------------------------------------

extern "C" void kernel_launch(void* const* d_in, const int* in_sizes, int n_in,
                              void* d_out, int out_size, void* d_ws, size_t ws_size,
                              hipStream_t stream) {
    const float* q    = (const float*)d_in[0];
    const float* k    = (const float*)d_in[1];
    const float* v    = (const float*)d_in[2];
    const int*   mask = (const int*)d_in[3];

    float* ctx  = (float*)d_out;                       // (B,T,D)
    float* dist = ctx + (size_t)B_ * T_ * D_;          // (B,T,T)

    __bf16* qp = (__bf16*)d_ws;                        // 8 MB fragment-major Q
    __bf16* kp = qp + (size_t)B_ * T_ * D_;            // 8 MB fragment-major K
    __bf16* vp = kp + (size_t)B_ * T_ * D_;            // 8 MB fragment-major V

    const int n = B_ * T_ * D_;                        // 4,194,304 elements
    pack_frag_qk_kernel<<<n / 256, 256, 0, stream>>>(q, k, qp, kp);
    pack_frag_v_kernel<<<B_ * H_ * 64, 256, 0, stream>>>(v, vp);

    dim3 ga(T_ / QT_, B_);
    attn_kernel<<<ga, 256, 0, stream>>>(qp, kp, vp, mask, ctx, dist);
}
